// LSTMEncoder_55946243998096
// MI455X (gfx1250) — compile-verified, matrix operand order checked by reading, not run on hardware
//
#include <hip/hip_runtime.h>
#include <hip/hip_bf16.h>
#include <cstddef>
#include <cstdint>

// ---------------------------------------------------------------------------
// LSTM encoder, MI455X (gfx1250, wave32, WMMA bf16 16x16x32, f32 accum)
//
//   pre[t,n,:] = [xs[t,n,:6], 1] @ [Wc | bc]^T,  Wc = W_ih @ W_emb (1024x6),
//                                               bc = W_ih @ b_emb + b_ih + b_hh
//   gates = pre[t] + h @ W_hh^T ; i,f,g,o ; c,h update ; out h.
//
//   prep_pack_whh : W_hh -> bf16 B-fragment layout (512 KB, L2-resident)
//   prep_pre      : [Wc | bc] packed as one K=32 B-tile (K=0..5 data, K=6 bias)
//   lstm_rec      : 64 WGs x 32 rows (2 row-tiles); h in LDS bf16, c in regs;
//                   144 WMMAs / wave / timestep; B fragments streamed from L2
//                   with explicit one-phase-ahead double buffering (8 loads
//                   = 8 KB per wave in flight) to hide L2 latency inside the wave.
// ---------------------------------------------------------------------------

#define AS1 __attribute__((address_space(1)))

typedef __attribute__((ext_vector_type(16))) __bf16 v16bf;
typedef __attribute__((ext_vector_type(8)))  __bf16 v8bf;
typedef __attribute__((ext_vector_type(8)))  float  v8f;
typedef const v16bf AS1* gv16bf_p;

#define T_STEPS    128
#define HDIM       256
#define GDIM       1024   // 4*HDIM
#define WG_ROWS    32     // rows per workgroup (2 row-tiles of 16)

// K index of fragment element j (0..15) in lane l, for 16-bit A (16x32, M=l&15)
// and B (32x16, N=l&15) fragments, per CDNA5 ISA 7.12.2.
__host__ __device__ inline int kof(int l, int j) {
    int v = j >> 1, h = j & 1;
    return ((v >> 2) << 4) + ((l >> 4) << 3) + ((v & 3) << 1) + h;
}

__device__ inline v8f wmma_bf16(v16bf a, v16bf b, v8f c) {
    return __builtin_amdgcn_wmma_f32_16x16x32_bf16(
        false, a, false, b, (short)0, c, false, false);
}

__device__ inline float sigm(float x) { return 1.0f / (1.0f + __expf(-x)); }
__device__ inline float tanh_fast(float x) {
    float e = __expf(2.0f * x);
    return 1.0f - 2.0f / (e + 1.0f);   // saturates cleanly at +/-1
}

// ---- prep: pack W_hh into B-fragment order: Bh[kt][ct][lane][j] (bf16) ----
__global__ __launch_bounds__(256) void prep_pack_whh(
    const float* __restrict__ W_hh, __bf16* __restrict__ Bh) {
    int idx = blockIdx.x * 256 + threadIdx.x;      // 0 .. 262143
    int kt  = idx >> 15;
    int rem = idx & 32767;
    int ct  = rem >> 9;
    int l   = (rem >> 4) & 31;
    int j   = rem & 15;
    int K   = kt * 32 + kof(l, j);
    int col = ct * 16 + (l & 15);
    Bh[idx] = (__bf16)W_hh[col * HDIM + K];
}

// ---- prep: composed input projection [Wc | bc] as one B-fragment tile ----
__global__ __launch_bounds__(256) void prep_pre(
    const float* __restrict__ W_emb, const float* __restrict__ b_emb,
    const float* __restrict__ W_ih,  const float* __restrict__ b_ih,
    const float* __restrict__ b_hh,
    __bf16* __restrict__ Bpre) {
    int idx = blockIdx.x * 256 + threadIdx.x;      // 0 .. 32767
    int ct  = idx >> 9;
    int l   = (idx >> 4) & 31;
    int j   = idx & 15;
    int K   = kof(l, j);
    float v = 0.0f;
    if (l < 16) {
        int col = ct * 16 + (l & 15);
        const float* wr = W_ih + (size_t)col * HDIM;
        if (K < 6) {                         // Wc = W_ih @ W_emb
            float s = 0.0f;
            for (int e = 0; e < HDIM; ++e) s += wr[e] * W_emb[e * 6 + K];
            v = s;
        } else if (K == 6) {                 // bias feature column
            float s = 0.0f;
            for (int e = 0; e < HDIM; ++e) s += wr[e] * b_emb[e];
            v = s + b_ih[col] + b_hh[col];
        }
    }
    Bpre[idx] = (__bf16)v;
}

// ---- main recurrence: one WG owns 32 batch rows for all 128 timesteps ----
__global__ __launch_bounds__(256) void lstm_rec(
    const float* __restrict__ inp,           // [2048][128][6]
    const __bf16* __restrict__ Bh,           // [8][64][32][16] bf16
    const __bf16* __restrict__ Bpre,         // [64][32][16] bf16
    float* __restrict__ out) {               // [2048][128][256]
    __shared__ __bf16 h_lds[WG_ROWS * HDIM]; // 16 KB, row-major [row][hcol]

    const int tid = threadIdx.x;
    const int w   = tid >> 5;                // wave 0..7
    const int l   = tid & 31;
    const int lm  = l & 15, lh = l >> 4;
    const int n0  = blockIdx.x * WG_ROWS;

    for (int i = tid; i < WG_ROWS * HDIM; i += 256) h_lds[i] = (__bf16)0.0f;

    v8f zeroc;                               // loop-invariant zero C operand
#pragma unroll
    for (int e = 0; e < 8; ++e) zeroc[e] = 0.0f;

    v8f cst[2][2];                           // c state [rt][s], C-frag layout
#pragma unroll
    for (int rt = 0; rt < 2; ++rt)
#pragma unroll
        for (int s = 0; s < 2; ++s)
#pragma unroll
            for (int e = 0; e < 8; ++e) cst[rt][s][e] = 0.0f;

    __syncthreads();

    for (int t = 0; t < T_STEPS; ++t) {
        // Launder fragment bases: per-iteration opaque value blocks LICM
        // (no 576-reg hoist + scratch spill); inttoptr into AS1 keeps loads
        // on the GLOBAL path (LOADcnt only), not FLAT (LOADcnt+DScnt).
        uintptr_t bh_u = (uintptr_t)Bh;
        uintptr_t bp_u = (uintptr_t)Bpre;
        asm volatile("" : "+s"(bh_u));
        asm volatile("" : "+s"(bp_u));
        gv16bf_p Bh_v = (gv16bf_p)bh_u;
        gv16bf_p Bp_v = (gv16bf_p)bp_u;

        // --- phase 0 B fragments ([Wc|bc] tile), issued first ---
        v16bf bbuf[8], bnext[8];
#pragma unroll
        for (int q = 0; q < 8; ++q) {        // q = g*2 + s
            int ct = (q >> 1) * 16 + (q & 1) * 8 + w;
            bbuf[q] = Bp_v[ct * 32 + l];
        }

        // --- A-fragments of xs[t] per row-tile (K<6 data, K=6 -> 1.0) ---
        v16bf axs[2];
#pragma unroll
        for (int rt = 0; rt < 2; ++rt) {
            const float* xp =
                inp + ((size_t)(n0 + rt * 16 + lm) * T_STEPS + t) * 6;
#pragma unroll
            for (int j = 0; j < 16; ++j) {
                float v = 0.0f;
                if (l < 16) {
                    if (j < 6) v = xp[j];
                    else if (j == 6) v = 1.0f;   // bias feature
                }
                axs[rt][j] = (__bf16)v;
            }
        }

        // --- 9 phases: p=0 input projection (C=0), p=1..8 kt=p-1 of h@W_hh^T.
        //     Phase p issues phase p+1's 8 B loads before its 16 WMMAs. ---
        v8f acc[2][4][2];
#pragma unroll
        for (int p = 0; p < 9; ++p) {
            if (p < 8) {                     // prefetch next phase's B frags
#pragma unroll
                for (int q = 0; q < 8; ++q) {
                    int ct = (q >> 1) * 16 + (q & 1) * 8 + w;
                    bnext[q] = Bh_v[(p * 64 + ct) * 32 + l];
                }
            }
            v16bf a0, a1;
            if (p == 0) {
                a0 = axs[0];
                a1 = axs[1];
            } else {
                int kt = p - 1;
#pragma unroll
                for (int rt = 0; rt < 2; ++rt) {
                    int base = (rt * 16 + lm) * HDIM + kt * 32 + lh * 8;
                    v8bf lo = *(const v8bf*)&h_lds[base];
                    v8bf hi = *(const v8bf*)&h_lds[base + 16];
#pragma unroll
                    for (int j = 0; j < 8; ++j) {
                        if (rt == 0) { a0[j] = lo[j]; a0[8 + j] = hi[j]; }
                        else         { a1[j] = lo[j]; a1[8 + j] = hi[j]; }
                    }
                }
            }
#pragma unroll
            for (int q = 0; q < 8; ++q) {
                int g = q >> 1, s = q & 1;
                if (p == 0) {
                    acc[0][g][s] = wmma_bf16(a0, bbuf[q], zeroc);
                    acc[1][g][s] = wmma_bf16(a1, bbuf[q], zeroc);
                } else {
                    acc[0][g][s] = wmma_bf16(a0, bbuf[q], acc[0][g][s]);
                    acc[1][g][s] = wmma_bf16(a1, bbuf[q], acc[1][g][s]);
                }
            }
#pragma unroll
            for (int q = 0; q < 8; ++q) bbuf[q] = bnext[q];  // renamed away
        }

        __syncthreads();   // all waves done reading h_lds for step t

        // --- gates -> c,h update; wave w owns hidden cols {16w.., 128+16w..} ---
#pragma unroll
        for (int rt = 0; rt < 2; ++rt)
#pragma unroll
            for (int s = 0; s < 2; ++s)
#pragma unroll
                for (int e = 0; e < 8; ++e) {
                    float iv = sigm(acc[rt][0][s][e]);
                    float fv = sigm(acc[rt][1][s][e]);
                    float gv = tanh_fast(acc[rt][2][s][e]);
                    float ov = sigm(acc[rt][3][s][e]);
                    float cn = fv * cst[rt][s][e] + iv * gv;
                    cst[rt][s][e] = cn;
                    float hv = ov * tanh_fast(cn);
                    int row = rt * 16 + e + lh * 8;
                    int col = (s * 8 + w) * 16 + lm;
                    h_lds[row * HDIM + col] = (__bf16)hv;
                    out[((size_t)(n0 + row) * T_STEPS + t) * HDIM + col] = hv;
                }

        __syncthreads();   // h(t+1) visible before next step's A loads
    }
}

// ---------------------------------------------------------------------------
extern "C" void kernel_launch(void* const* d_in, const int* in_sizes, int n_in,
                              void* d_out, int out_size, void* d_ws, size_t ws_size,
                              hipStream_t stream) {
    const float* inp   = (const float*)d_in[0];  // [32,64,128,6]
    const float* W_emb = (const float*)d_in[1];  // [256,6]
    const float* b_emb = (const float*)d_in[2];  // [256]
    const float* W_ih  = (const float*)d_in[3];  // [1024,256]
    const float* W_hh  = (const float*)d_in[4];  // [1024,256]
    const float* b_ih  = (const float*)d_in[5];  // [1024]
    const float* b_hh  = (const float*)d_in[6];  // [1024]
    float* out = (float*)d_out;

    char* ws = (char*)d_ws;
    __bf16* Bh   = (__bf16*)ws;                          // 524288 B
    __bf16* Bpre = (__bf16*)(ws + 524288);               //  65536 B

    prep_pack_whh<<<1024, 256, 0, stream>>>(W_hh, Bh);
    prep_pre<<<128, 256, 0, stream>>>(W_emb, b_emb, W_ih, b_ih, b_hh, Bpre);
    lstm_rec<<<64, 256, 0, stream>>>(inp, Bh, Bpre, out);
}
